// _LightGCN_90555090468953
// MI455X (gfx1250) — matrix operand order repaired and checked by reading, not compiled, not run on hardware
//
#include <hip/hip_runtime.h>
#include <hip/hip_bf16.h>

// ---------------- problem constants (match reference) ----------------
#define NUM_USERS 60000
#define NUM_ITEMS 40000
#define NNODES    100000
#define DIM       64
#define BATCH     2048
#define NLAYERS   3
#define SSL_TEMP  0.2f

typedef __attribute__((ext_vector_type(16))) _Float16 v16h;
typedef __attribute__((ext_vector_type(8)))  float    v8f;
typedef __attribute__((ext_vector_type(4)))  float    v4f;   // native vector (NT-store OK)

// Native CDNA5 f32 scatter-add: guarantees global_atomic_add_f32 (no CAS loop).
// Non-returning form -> tracked by STOREcnt; S_ENDPGM's implicit wait-idle plus
// stream ordering makes results visible to the next kernel through L2.
__device__ __forceinline__ void atomic_add_f32(float* p, float v) {
    asm volatile("global_atomic_add_f32 %0, %1, off"
                 :: "v"(p), "v"(v)
                 : "memory");
}

__device__ __forceinline__ float wave_reduce_sum(float v) {
    #pragma unroll
    for (int m = 16; m; m >>= 1) v += __shfl_xor(v, m, 32);
    return v;
}

// ---------------- GCN elementwise kernels ----------------
__global__ void k_init_ego(const float* __restrict__ syn, const float* __restrict__ herb,
                           float* __restrict__ ego, float* __restrict__ acc) {
    size_t i = (size_t)blockIdx.x * blockDim.x + threadIdx.x;
    const size_t nd = (size_t)NNODES * DIM;
    if (i >= nd) return;
    const size_t ub = (size_t)NUM_USERS * DIM;
    float v = (i < ub) ? syn[i] : herb[i - ub];
    ego[i] = v; acc[i] = v;
}

__global__ void k_zero(float* __restrict__ p, size_t n) {
    size_t i = (size_t)blockIdx.x * blockDim.x + threadIdx.x;
    if (i < n) p[i] = 0.0f;
}

// edge-parallel COO SpMM: 8 threads per edge, 8 floats each (vector gathers)
__global__ void k_spmm(const int* __restrict__ row, const int* __restrict__ col,
                       const float* __restrict__ val, const float* __restrict__ x,
                       float* __restrict__ y, int ne) {
    size_t t = (size_t)blockIdx.x * blockDim.x + threadIdx.x;
    int e = (int)(t >> 3);
    if (e >= ne) return;
    int dk = ((int)t & 7) * 8;
    int r = row[e], c = col[e];
    float v = val[e];
    const v4f* xs = (const v4f*)(x + (size_t)c * DIM + dk);
    v4f x0 = xs[0], x1 = xs[1];
    float* yd = y + (size_t)r * DIM + dk;
    atomic_add_f32(yd + 0, v * x0.x); atomic_add_f32(yd + 1, v * x0.y);
    atomic_add_f32(yd + 2, v * x0.z); atomic_add_f32(yd + 3, v * x0.w);
    atomic_add_f32(yd + 4, v * x1.x); atomic_add_f32(yd + 5, v * x1.y);
    atomic_add_f32(yd + 6, v * x1.z); atomic_add_f32(yd + 7, v * x1.w);
}

__global__ void k_update(float* __restrict__ ego, const float* __restrict__ neigh,
                         float* __restrict__ acc, const float* __restrict__ alpha) {
    size_t i = (size_t)blockIdx.x * blockDim.x + threadIdx.x;
    const size_t nd = (size_t)NNODES * DIM;
    if (i >= nd) return;
    float a = 1.0f / (1.0f + __expf(-alpha[0]));
    float e = ego[i];
    float v = fmaxf(a * neigh[i] + (1.0f - a) * e, 0.0f);
    ego[i] = v; acc[i] += v;
}

__global__ void k_avg_store(const float* __restrict__ acc, float* __restrict__ out) {
    size_t i = (size_t)blockIdx.x * blockDim.x + threadIdx.x;
    const size_t nd = (size_t)NNODES * DIM;
    if (i < nd) out[i] = acc[i] * 0.25f;
}

// avg + row l2-normalize; writes f32 and f16 copies. one wave per row.
__global__ void k_avg_l2norm(const float* __restrict__ acc, float* __restrict__ z,
                             _Float16* __restrict__ zh) {
    int r = blockIdx.x * (blockDim.x >> 5) + (threadIdx.x >> 5);
    int lane = threadIdx.x & 31;
    if (r >= NNODES) return;
    const float* a = acc + (size_t)r * DIM;
    float x0 = a[lane] * 0.25f, x1 = a[lane + 32] * 0.25f;
    float ss = wave_reduce_sum(x0 * x0 + x1 * x1);
    float inv = 1.0f / fmaxf(sqrtf(ss), 1e-12f);
    x0 *= inv; x1 *= inv;
    size_t base = (size_t)r * DIM;
    z[base + lane] = x0; z[base + lane + 32] = x1;
    zh[base + lane] = (_Float16)x0; zh[base + lane + 32] = (_Float16)x1;
}

// ---------------- batch kernels ----------------
__global__ void k_sup(const float* __restrict__ uemb, const float* __restrict__ iemb,
                      const int* __restrict__ users, const int* __restrict__ items,
                      const int* __restrict__ negs, float* __restrict__ out) {
    int b = blockIdx.x * (blockDim.x >> 5) + (threadIdx.x >> 5);
    int lane = threadIdx.x & 31;
    if (b >= BATCH) return;
    const float* ue = uemb + (size_t)users[b] * DIM;
    const float* ie = iemb + (size_t)items[b] * DIM;
    const float* ne = iemb + (size_t)negs[b]  * DIM;
    float d = ue[lane] * (ie[lane] - ne[lane])
            + ue[lane + 32] * (ie[lane + 32] - ne[lane + 32]);
    d = wave_reduce_sum(d);
    if (lane == 0) out[b] = d;
}

// gather batch rows of z1h/z2h into packed f16 A-matrices, and compute pos dots (f32)
__global__ void k_gather(const float* __restrict__ z1, const float* __restrict__ z2,
                         const _Float16* __restrict__ z1h, const _Float16* __restrict__ z2h,
                         const int* __restrict__ users, const int* __restrict__ items,
                         _Float16* __restrict__ ue1h, _Float16* __restrict__ ue2h,
                         _Float16* __restrict__ ie1h, _Float16* __restrict__ ie2h,
                         float* __restrict__ pos_u, float* __restrict__ pos_i) {
    int w = blockIdx.x * (blockDim.x >> 5) + (threadIdx.x >> 5);
    int lane = threadIdx.x & 31;
    if (w >= 2 * BATCH) return;
    int b, node; _Float16 *d1, *d2; float* pos;
    if (w < BATCH) {
        b = w; node = users[b];
        d1 = ue1h + (size_t)b * DIM; d2 = ue2h + (size_t)b * DIM; pos = pos_u;
    } else {
        b = w - BATCH; node = NUM_USERS + items[b];
        d1 = ie1h + (size_t)b * DIM; d2 = ie2h + (size_t)b * DIM; pos = pos_i;
    }
    size_t base = (size_t)node * DIM;
    d1[lane] = z1h[base + lane]; d1[lane + 32] = z1h[base + lane + 32];
    d2[lane] = z2h[base + lane]; d2[lane + 32] = z2h[base + lane + 32];
    float d = z1[base + lane] * z2[base + lane]
            + z1[base + lane + 32] * z2[base + lane + 32];
    d = wave_reduce_sum(d);
    if (lane == 0) pos[b] = d;
}

// ---------------- WMMA GEMM: C[m,n] = scale * dot(A[m,:], B[n,:]) - rowsub[m] ----------------
// A: [M,64] f16 row-major, B: [ncols,64] f16 row-major (i.e. C = A * B^T), K = 64.
// rowsub must always be a valid pointer (use a zeroed vector for "no bias").
// block = (32,4): 4 waves; each wave computes a 16(M) x 64(N) strip (4 tiles, A reuse).
// Epilogue stages the strip in LDS, then writes coalesced 256B row segments (b128, NT).
__global__ void k_gemm_bt(const _Float16* __restrict__ A, const _Float16* __restrict__ Bm,
                          const float* __restrict__ rowsub, float* __restrict__ C,
                          int ncols, float scale) {
    __shared__ float tile[4][16][68];   // stride 68: halves hit disjoint bank ranges
    const int lane = threadIdx.x;
    const int wv = threadIdx.y;
    const int half = lane >> 4, l = lane & 15;
    const int mbase = blockIdx.y * 16;

    union U16 { v4f f[2]; v16h h; };
    // 16-bit A fragment 16x32 (ISA layout): lane-half selects interleaved K-octets
    const v4f* arow = (const v4f*)(A + (size_t)(mbase + l) * DIM); // 8 x v4f per row
    U16 a0, a1;
    a0.f[0] = arow[half];     a0.f[1] = arow[2 + half];   // K = 0..31
    a1.f[0] = arow[4 + half]; a1.f[1] = arow[6 + half];   // K = 32..63

    // row-bias values for the 8 rows this lane-half covers (branch-free epilogue)
    float subv[8];
    #pragma unroll
    for (int j = 0; j < 8; ++j) subv[j] = rowsub[mbase + half * 8 + j];

    const int ntg = blockIdx.x * 16 + wv * 4;             // first N-tile for this wave
    #pragma unroll
    for (int nt = 0; nt < 4; ++nt) {
        int ntile = ntg + nt;
        if (ntile * 16 >= ncols) break;                   // wave-uniform guard
        // B fragment 32x16: lane-half selects interleaved K-octets of column n (matches A)
        const v4f* brow = (const v4f*)(Bm + (size_t)(ntile * 16 + l) * DIM);
        U16 b0, b1;
        b0.f[0] = brow[half];     b0.f[1] = brow[2 + half];
        b1.f[0] = brow[4 + half]; b1.f[1] = brow[6 + half];

        v8f c = {};
        c = __builtin_amdgcn_wmma_f32_16x16x32_f16(false, a0.h, false, b0.h,
                                                   (short)0, c, false, false);
        c = __builtin_amdgcn_wmma_f32_16x16x32_f16(false, a1.h, false, b1.h,
                                                   (short)0, c, false, false);
        #pragma unroll
        for (int j = 0; j < 8; ++j)                       // C/D layout: VGPR j -> M=j+8*half
            tile[wv][half * 8 + j][nt * 16 + l] = c[j] * scale - subv[j];
    }
    __syncthreads();

    // coalesced writeback: each half-wave emits one contiguous 256B row segment per step
    #pragma unroll
    for (int p = 0; p < 8; ++p) {
        int r = p * 2 + half;                             // rows 0..15 of the strip
        int n0 = ntg * 16 + l * 4;                        // float4 column start
        if (n0 < ncols) {
            v4f v = *(const v4f*)&tile[wv][r][l * 4];
            __builtin_nontemporal_store(v, (v4f*)&C[(size_t)(mbase + r) * ncols + n0]);
        }
    }
}

// ---------------- logsumexp over rows of [BATCH x BATCH] ----------------
__global__ void k_lse(const float* __restrict__ S, float* __restrict__ ttl) {
    __shared__ float red[256];
    int b = blockIdx.x, tid = threadIdx.x;
    const float* r = S + (size_t)b * BATCH;
    float mx = -3.0e38f;
    for (int j = tid; j < BATCH; j += 256) mx = fmaxf(mx, r[j]);
    red[tid] = mx; __syncthreads();
    for (int s = 128; s; s >>= 1) {
        if (tid < s) red[tid] = fmaxf(red[tid], red[tid + s]);
        __syncthreads();
    }
    mx = red[0]; __syncthreads();
    float sm = 0.0f;
    for (int j = tid; j < BATCH; j += 256) sm += __expf(r[j] - mx);
    red[tid] = sm; __syncthreads();
    for (int s = 128; s; s >>= 1) {
        if (tid < s) red[tid] += red[tid + s];
        __syncthreads();
    }
    if (tid == 0) ttl[b] = __logf(red[0]) + mx;
}

__global__ void k_loss(const float* __restrict__ ttl_u, const float* __restrict__ ttl_i,
                       const float* __restrict__ pos_u, const float* __restrict__ pos_i,
                       float* __restrict__ out) {
    __shared__ float red[256];
    int tid = threadIdx.x;
    const float it = 1.0f / SSL_TEMP;
    float s = 0.0f;
    for (int b = tid; b < BATCH; b += 256)
        s += (ttl_u[b] - pos_u[b] * it) + (ttl_i[b] - pos_i[b] * it);
    red[tid] = s; __syncthreads();
    for (int k = 128; k; k >>= 1) {
        if (tid < k) red[tid] += red[tid + k];
        __syncthreads();
    }
    if (tid == 0) out[0] = red[0];
}

// ---------------- host driver ----------------
extern "C" void kernel_launch(void* const* d_in, const int* in_sizes, int n_in,
                              void* d_out, int out_size, void* d_ws, size_t ws_size,
                              hipStream_t stream) {
    const float* syn  = (const float*)d_in[0];
    const float* herb = (const float*)d_in[1];
    const int*   a_row = (const int*)d_in[2];
    const int*   a_col = (const int*)d_in[3];
    const float* a_val = (const float*)d_in[4];
    const int*   s1_row = (const int*)d_in[5];
    const int*   s1_col = (const int*)d_in[6];
    const float* s1_val = (const float*)d_in[7];
    const int*   s2_row = (const int*)d_in[8];
    const int*   s2_col = (const int*)d_in[9];
    const float* s2_val = (const float*)d_in[10];
    const int*   users = (const int*)d_in[11];
    const int*   items = (const int*)d_in[12];
    const int*   negs  = (const int*)d_in[13];
    const float* alpha = (const float*)d_in[14];
    const int nedges = in_sizes[2];

    float* out = (float*)d_out;
    const size_t OFF_SUP  = 0;
    const size_t OFF_SSLU = BATCH;
    const size_t OFF_SSLI = OFF_SSLU + (size_t)BATCH * NUM_USERS;
    const size_t OFF_UEMB = OFF_SSLI + (size_t)BATCH * NUM_ITEMS;
    const size_t OFF_LOSS = OFF_UEMB + (size_t)NNODES * DIM;

    const size_t ND = (size_t)NNODES * DIM;
    float*    ego   = (float*)d_ws;
    float*    neigh = ego + ND;
    float*    acc   = neigh + ND;
    float*    z1    = acc + ND;
    float*    z2    = z1 + ND;
    _Float16* z1h   = (_Float16*)(z2 + ND);
    _Float16* z2h   = z1h + ND;
    _Float16* ue1h  = z2h + ND;
    _Float16* ue2h  = ue1h + (size_t)BATCH * DIM;
    _Float16* ie1h  = ue2h + (size_t)BATCH * DIM;
    _Float16* ie2h  = ie1h + (size_t)BATCH * DIM;
    float*    pos_u = (float*)(ie2h + (size_t)BATCH * DIM);
    float*    pos_i = pos_u + BATCH;
    float*    ttl_u = pos_i + BATCH;
    float*    ttl_i = ttl_u + BATCH;
    float*    Sbuf  = ttl_i + BATCH;            // BATCH*BATCH f32
    float*    zerob = Sbuf + (size_t)BATCH * BATCH;  // BATCH zeros (no-bias epilogue)

    const int TPB = 256;
    const int ndBlocks = (int)((ND + TPB - 1) / TPB);
    const size_t spmmThreads = (size_t)nedges * 8;
    const int spmmBlocks = (int)((spmmThreads + TPB - 1) / TPB);

    auto run_gcn = [&](const int* row, const int* col, const float* val) {
        k_init_ego<<<ndBlocks, TPB, 0, stream>>>(syn, herb, ego, acc);
        for (int l = 0; l < NLAYERS; ++l) {
            k_zero<<<ndBlocks, TPB, 0, stream>>>(neigh, ND);
            k_spmm<<<spmmBlocks, TPB, 0, stream>>>(row, col, val, ego, neigh, nedges);
            k_update<<<ndBlocks, TPB, 0, stream>>>(ego, neigh, acc, alpha);
        }
    };

    // full graph -> u_emb / i_emb (contiguous in d_out)
    run_gcn(a_row, a_col, a_val);
    k_avg_store<<<ndBlocks, TPB, 0, stream>>>(acc, out + OFF_UEMB);
    // subgraph views -> normalized z1/z2 (+ f16 mirrors)
    run_gcn(s1_row, s1_col, s1_val);
    k_avg_l2norm<<<(NNODES + 7) / 8, 256, 0, stream>>>(acc, z1, z1h);
    run_gcn(s2_row, s2_col, s2_val);
    k_avg_l2norm<<<(NNODES + 7) / 8, 256, 0, stream>>>(acc, z2, z2h);

    // supervised logits + batch gathers
    k_sup<<<(BATCH + 7) / 8, 256, 0, stream>>>(out + OFF_UEMB,
                                               out + OFF_UEMB + (size_t)NUM_USERS * DIM,
                                               users, items, negs, out + OFF_SUP);
    k_gather<<<(2 * BATCH + 7) / 8, 256, 0, stream>>>(z1, z2, z1h, z2h, users, items,
                                                      ue1h, ue2h, ie1h, ie2h, pos_u, pos_i);
    k_zero<<<(BATCH + TPB - 1) / TPB, TPB, 0, stream>>>(zerob, (size_t)BATCH);

    // WMMA GEMMs (C = A * B^T). block covers 16 M-rows x 256 N-cols.
    dim3 blk(32, 4);
    k_gemm_bt<<<dim3((NUM_USERS + 255) / 256, BATCH / 16), blk, 0, stream>>>(
        ue1h, z2h, pos_u, out + OFF_SSLU, NUM_USERS, 1.0f);
    k_gemm_bt<<<dim3((NUM_ITEMS + 255) / 256, BATCH / 16), blk, 0, stream>>>(
        ie1h, z2h + (size_t)NUM_USERS * DIM, pos_i, out + OFF_SSLI, NUM_ITEMS, 1.0f);

    // InfoNCE: scores/temp -> logsumexp -> loss
    k_gemm_bt<<<dim3(BATCH / 256, BATCH / 16), blk, 0, stream>>>(
        ue1h, ue2h, zerob, Sbuf, BATCH, 1.0f / SSL_TEMP);
    k_lse<<<BATCH, 256, 0, stream>>>(Sbuf, ttl_u);
    k_gemm_bt<<<dim3(BATCH / 256, BATCH / 16), blk, 0, stream>>>(
        ie1h, ie2h, zerob, Sbuf, BATCH, 1.0f / SSL_TEMP);
    k_lse<<<BATCH, 256, 0, stream>>>(Sbuf, ttl_i);
    k_loss<<<1, 256, 0, stream>>>(ttl_u, ttl_i, pos_u, pos_i, out + OFF_LOSS);
}
